// Block_10230612099678
// MI455X (gfx1250) — compile-verified, matrix operand order checked by reading, not run on hardware
//
#include <hip/hip_runtime.h>

typedef __attribute__((ext_vector_type(16))) _Float16 v16h;
typedef __attribute__((ext_vector_type(8)))  float    v8f;
typedef __attribute__((ext_vector_type(2)))  _Float16 h2;

#define S_LEN 4096
#define EMB   768
#define NH    12
#define HD    64
#define WIN_  256
#define QKV_N 2304
#define NTOK  8192   // B * S

// ---------------------------------------------------------------------------
// WMMA helpers (CDNA5 gfx1250, wave32, 16x16x32 f16 -> f32)
// ---------------------------------------------------------------------------
__device__ __forceinline__ v8f wmma16(v16h a, v16h b, v8f c) {
  return __builtin_amdgcn_wmma_f32_16x16x32_f16(false, a, false, b, (short)0, c,
                                                false, false);
}

// Gather one 16-bit A/B fragment (16 halfs / lane) from a k-contiguous row.
// ISA 7.12.2: lane row = lane&15, hi = lane>>4; halfs [8*hi, 8*hi+8) and
// [16+8*hi, 16+8*hi+8) -> two contiguous 16B chunks (lower to 2x ds_load_b128).
__device__ __forceinline__ v16h load_frag(const _Float16* rowbase, int hi) {
  v16h f;
#pragma unroll
  for (int p = 0; p < 8; ++p) {
    int k = ((p < 4) ? (2 * p) : (16 + 2 * (p - 4))) + hi * 8;
    h2 t = *(const h2*)(rowbase + k);
    f[2 * p]     = t[0];
    f[2 * p + 1] = t[1];
  }
  return f;
}

// ---------------------------------------------------------------------------
// Weight prep: f32 (K x N) -> f16 transposed (N x K), 32x32 tiles via LDS
// ---------------------------------------------------------------------------
__global__ __launch_bounds__(256) void cvt_transpose_f16(
    const float* __restrict__ in, _Float16* __restrict__ out, int Kd, int Nd) {
  __shared__ float tile[32][33];
  const int n0 = blockIdx.x * 32, k0 = blockIdx.y * 32;
  const int tx = threadIdx.x & 31, ty = threadIdx.x >> 5;  // 32 x 8
#pragma unroll
  for (int i = ty; i < 32; i += 8)
    tile[i][tx] = in[(size_t)(k0 + i) * Nd + n0 + tx];
  __syncthreads();
#pragma unroll
  for (int i = ty; i < 32; i += 8)
    out[(size_t)(n0 + i) * Kd + k0 + tx] = (_Float16)tile[tx][i];
}

// ---------------------------------------------------------------------------
// LayerNorm over E=768: one block (256 threads, 8 waves) per row, f16 out
// ---------------------------------------------------------------------------
__global__ __launch_bounds__(256) void ln_kernel(const float* __restrict__ x,
                                                 const float* __restrict__ g,
                                                 const float* __restrict__ b,
                                                 _Float16* __restrict__ out) {
  __shared__ float red[18];
  const int row = blockIdx.x;
  const int t = threadIdx.x;
  const float* xr = x + (size_t)row * EMB;
  float v0 = xr[t], v1 = xr[t + 256], v2 = xr[t + 512];
  float s  = v0 + v1 + v2;
  float s2 = v0 * v0 + v1 * v1 + v2 * v2;
#pragma unroll
  for (int off = 16; off > 0; off >>= 1) {
    s  += __shfl_down(s,  off, 32);
    s2 += __shfl_down(s2, off, 32);
  }
  int lane = t & 31, wid = t >> 5;
  if (lane == 0) { red[wid] = s; red[8 + wid] = s2; }
  __syncthreads();
  if (t == 0) {
    float S = 0.f, S2 = 0.f;
    for (int i = 0; i < 8; ++i) { S += red[i]; S2 += red[8 + i]; }
    float m = S / (float)EMB;
    red[16] = m;
    red[17] = rsqrtf(S2 / (float)EMB - m * m + 1e-5f);
  }
  __syncthreads();
  float m = red[16], inv = red[17];
  _Float16* orow = out + (size_t)row * EMB;
  orow[t]       = (_Float16)((v0 - m) * inv * g[t]       + b[t]);
  orow[t + 256] = (_Float16)((v1 - m) * inv * g[t + 256] + b[t + 256]);
  orow[t + 512] = (_Float16)((v2 - m) * inv * g[t + 512] + b[t + 512]);
}

// ---------------------------------------------------------------------------
// f16 WMMA GEMM with async global->LDS double buffering.
// C(MxN) = A(MxK) * Bt(NxK)^T + bias [+gelu] [+residual]
// Block tile 256x128, BK=32, 256 threads = 8 waves (4x2), wave tile 64x64.
// One barrier per K-step; next tile streams via GLOBAL_LOAD_ASYNC_TO_LDS_B128
// while WMMAs consume the current buffer (ASYNCcnt-tracked).
// ---------------------------------------------------------------------------
#define BM  256
#define BN  128
#define BK  32
#define LDA 40   // padded half stride for ldsA rows (m, k)
#define LDB 40   // padded half stride for ldsB rows (n, k)

__device__ __forceinline__ void issue_async_tile(
    const _Float16* __restrict__ A, const _Float16* __restrict__ Bt,
    _Float16* lA, _Float16* lB, int m0, int n0, int K, int k0, int t) {
  // A tile: 256 rows x 64B; thread t copies row t (4 x b128).
  {
    unsigned lds = (unsigned)(unsigned long long)(lA + t * LDA);
    unsigned goff = (unsigned)(((unsigned)(m0 + t) * (unsigned)K + (unsigned)k0) * 2u);
    unsigned long long base = (unsigned long long)A;
    asm volatile(
        "global_load_async_to_lds_b128 %0, %1, %2\n\t"
        "global_load_async_to_lds_b128 %0, %1, %2 offset:16\n\t"
        "global_load_async_to_lds_b128 %0, %1, %2 offset:32\n\t"
        "global_load_async_to_lds_b128 %0, %1, %2 offset:48"
        :: "v"(lds), "v"(goff), "s"(base) : "memory");
  }
  // B tile: 128 rows x 64B; thread t copies half of row t>>1 (2 x b128).
  {
    int rb = t >> 1, cb = (t & 1) * 16;  // halfs
    unsigned lds = (unsigned)(unsigned long long)(lB + rb * LDB + cb);
    unsigned goff =
        (unsigned)((((unsigned)(n0 + rb) * (unsigned)K + (unsigned)k0 + (unsigned)cb)) * 2u);
    unsigned long long base = (unsigned long long)Bt;
    asm volatile(
        "global_load_async_to_lds_b128 %0, %1, %2\n\t"
        "global_load_async_to_lds_b128 %0, %1, %2 offset:16"
        :: "v"(lds), "v"(goff), "s"(base) : "memory");
  }
}

__global__ __launch_bounds__(256) void gemm_f16(
    const _Float16* __restrict__ A, const _Float16* __restrict__ Bt,
    const float* __restrict__ bias, const float* __restrict__ residual,
    float* __restrict__ outF, _Float16* __restrict__ outH,
    int M, int N, int K, int gelu_flag) {
  __shared__ _Float16 lA[2][BM * LDA];
  __shared__ _Float16 lB[2][BN * LDB];
  const int t = threadIdx.x;
  const int lane = t & 31, wid = t >> 5;
  const int lm = lane & 15, hi = lane >> 4;
  const int m0 = blockIdx.y * BM, n0 = blockIdx.x * BN;
  const int wm = (wid >> 1) * 64, wn = (wid & 1) * 64;

  v8f acc[4][4];
#pragma unroll
  for (int mi = 0; mi < 4; ++mi)
#pragma unroll
    for (int ni = 0; ni < 4; ++ni)
#pragma unroll
      for (int r = 0; r < 8; ++r) acc[mi][ni][r] = 0.f;

  const int KT = K / BK;
  issue_async_tile(A, Bt, lA[0], lB[0], m0, n0, K, 0, t);

  for (int kt = 0; kt < KT; ++kt) {
    asm volatile("s_wait_asynccnt 0x0" ::: "memory");  // my tile kt landed
    __syncthreads();  // all waves: tile kt visible; prev compute done
    if (kt + 1 < KT)
      issue_async_tile(A, Bt, lA[(kt + 1) & 1], lB[(kt + 1) & 1], m0, n0, K,
                       (kt + 1) * BK, t);
    const _Float16* cA = lA[kt & 1];
    const _Float16* cB = lB[kt & 1];
    // hoist all 8 fragment loads (16x ds_load_b128) so the scheduler can
    // interleave incremental s_wait_dscnt with the 16 WMMAs
    v16h af[4], bf[4];
#pragma unroll
    for (int mi = 0; mi < 4; ++mi)
      af[mi] = load_frag(cA + (wm + mi * 16 + lm) * LDA, hi);
#pragma unroll
    for (int ni = 0; ni < 4; ++ni)
      bf[ni] = load_frag(cB + (wn + ni * 16 + lm) * LDB, hi);
#pragma unroll
    for (int ni = 0; ni < 4; ++ni)
#pragma unroll
      for (int mi = 0; mi < 4; ++mi)
        acc[mi][ni] = wmma16(af[mi], bf[ni], acc[mi][ni]);
  }

  // Epilogue: bias [+gelu] [+residual]; f32 or f16 store
#pragma unroll
  for (int mi = 0; mi < 4; ++mi) {
#pragma unroll
    for (int ni = 0; ni < 4; ++ni) {
      int n = n0 + wn + ni * 16 + lm;
      float bv = bias[n];
#pragma unroll
      for (int r = 0; r < 8; ++r) {
        int m = m0 + wm + mi * 16 + r + 8 * hi;
        float v = acc[mi][ni][r] + bv;
        if (gelu_flag) v = 0.5f * v * (1.0f + erff(v * 0.70710678118654752f));
        if (residual) v += residual[(size_t)m * N + n];
        if (outF) outF[(size_t)m * N + n] = v;
        else      outH[(size_t)m * N + n] = (_Float16)v;
      }
    }
  }
}

// ---------------------------------------------------------------------------
// Sliding-window flash attention.
// Grid (S/64, H, B); 128 threads = 4 waves; wave owns 16 query rows.
// Key loop: 18 tiles of 32 keys covering [q0-256, q0+64+256).
// ---------------------------------------------------------------------------
#define LDK 66   // ldsK half stride [key][dim]
#define LDV 34   // ldsV half stride [dim][key] (transposed)
#define LDP 34   // ldsP half stride [row][key]

__global__ __launch_bounds__(128) void attn_kernel(const _Float16* __restrict__ qkv,
                                                   _Float16* __restrict__ outA) {
  __shared__ _Float16 ldsK[32 * LDK];
  __shared__ _Float16 ldsV[64 * LDV];
  __shared__ _Float16 ldsP[4 * 16 * LDP];
  const int t = threadIdx.x;
  const int lane = t & 31, wv = t >> 5;
  const int lm = lane & 15, hi = lane >> 4;
  const int q0 = blockIdx.x * 64;
  const int head = blockIdx.y, batch = blockIdx.z;
  const int tb = batch * S_LEN;
  const int ko = EMB + head * HD, vo = 2 * EMB + head * HD;

  // Q fragments (A layout: lane row = lm), loaded once, kept in VGPRs
  const _Float16* qrow =
      qkv + (size_t)(tb + q0 + wv * 16 + lm) * QKV_N + head * HD;
  v16h qf0 = load_frag(qrow, hi);
  v16h qf1 = load_frag(qrow + 32, hi);

  float mrow[8], lrow[8];
  v8f o[4];
#pragma unroll
  for (int r = 0; r < 8; ++r) { mrow[r] = -1e30f; lrow[r] = 0.f; }
#pragma unroll
  for (int s = 0; s < 4; ++s)
#pragma unroll
    for (int r = 0; r < 8; ++r) o[s][r] = 0.f;

  const int jstart = q0 - WIN_;
  for (int it = 0; it < 18; ++it) {
    const int jt = jstart + it * 32;
    __syncthreads();
    // cooperative K/V tile load: 32 keys x 64 dims (OOB keys -> zero)
    {
      int key = t >> 2, seg = t & 3;
      int j = jt + key;
      uint4 k0v = {0, 0, 0, 0}, k1v = {0, 0, 0, 0};
      uint4 v0v = {0, 0, 0, 0}, v1v = {0, 0, 0, 0};
      if (j >= 0 && j < S_LEN) {
        const uint4* kp = (const uint4*)(qkv + (size_t)(tb + j) * QKV_N + ko + seg * 16);
        k0v = kp[0]; k1v = kp[1];
        const uint4* vp = (const uint4*)(qkv + (size_t)(tb + j) * QKV_N + vo + seg * 16);
        v0v = vp[0]; v1v = vp[1];
      }
      {  // K: [key][dim], dword stores (stride 66 halfs is 4B aligned)
        const unsigned* s0 = (const unsigned*)&k0v;
        const unsigned* s1 = (const unsigned*)&k1v;
        unsigned* dst = (unsigned*)(ldsK + key * LDK + seg * 16);
#pragma unroll
        for (int i = 0; i < 4; ++i) { dst[i] = s0[i]; dst[4 + i] = s1[i]; }
      }
      {  // V: transposed [dim][key]
        const _Float16* e0 = (const _Float16*)&v0v;
        const _Float16* e1 = (const _Float16*)&v1v;
#pragma unroll
        for (int i = 0; i < 8; ++i) {
          ldsV[(seg * 16 + i) * LDV + key]     = e0[i];
          ldsV[(seg * 16 + 8 + i) * LDV + key] = e1[i];
        }
      }
    }
    __syncthreads();

    // scores: 16 queries x 32 keys (two 16-key C tiles), scale + band mask
    float sc[2][8];
    bool  vl[2][8];
#pragma unroll
    for (int gg = 0; gg < 2; ++gg) {
      v16h bk0 = load_frag(ldsK + (gg * 16 + lm) * LDK, hi);
      v16h bk1 = load_frag(ldsK + (gg * 16 + lm) * LDK + 32, hi);
      v8f c;
#pragma unroll
      for (int r = 0; r < 8; ++r) c[r] = 0.f;
      c = wmma16(qf0, bk0, c);
      c = wmma16(qf1, bk1, c);
      int j = jt + gg * 16 + lm;  // this lane's key column
#pragma unroll
      for (int r = 0; r < 8; ++r) {
        int iq = q0 + wv * 16 + r + 8 * hi;  // C layout row
        bool ok = (j >= 0) && (j < S_LEN) && (j >= iq - WIN_) && (j <= iq + WIN_);
        vl[gg][r] = ok;
        sc[gg][r] = ok ? c[r] * 0.125f : -1e30f;
      }
    }

    // online softmax update (row spread over 16 lanes of each half-wave)
#pragma unroll
    for (int r = 0; r < 8; ++r) {
      float tm = fmaxf(sc[0][r], sc[1][r]);
#pragma unroll
      for (int msk = 1; msk < 16; msk <<= 1)
        tm = fmaxf(tm, __shfl_xor(tm, msk, 32));
      float mnew = fmaxf(mrow[r], tm);
      float corr = __expf(mrow[r] - mnew);
      float p0 = vl[0][r] ? __expf(sc[0][r] - mnew) : 0.f;
      float p1 = vl[1][r] ? __expf(sc[1][r] - mnew) : 0.f;
      float rs = p0 + p1;
#pragma unroll
      for (int msk = 1; msk < 16; msk <<= 1)
        rs += __shfl_xor(rs, msk, 32);
      lrow[r] = lrow[r] * corr + rs;
      mrow[r] = mnew;
#pragma unroll
      for (int s = 0; s < 4; ++s) o[s][r] *= corr;
      // park P (C layout) into per-wave LDS for C->A re-layout
      _Float16* pw = ldsP + wv * (16 * LDP);
      pw[(r + 8 * hi) * LDP + lm]      = (_Float16)p0;
      pw[(r + 8 * hi) * LDP + 16 + lm] = (_Float16)p1;
    }

    // O += P(16x32) x V(32x64): one A frag, 4 B frags over dim subtiles
    v16h pf = load_frag(ldsP + wv * (16 * LDP) + lm * LDP, hi);
#pragma unroll
    for (int s = 0; s < 4; ++s) {
      v16h bvf = load_frag(ldsV + (s * 16 + lm) * LDV, hi);
      o[s] = wmma16(pf, bvf, o[s]);
    }
  }

  // normalize and write out (b,s,h,d) -> (tok, h*64+d), f16
#pragma unroll
  for (int s = 0; s < 4; ++s) {
#pragma unroll
    for (int r = 0; r < 8; ++r) {
      float v = o[s][r] / lrow[r];
      outA[(size_t)(tb + q0 + wv * 16 + r + 8 * hi) * EMB + head * HD + s * 16 + lm] =
          (_Float16)v;
    }
  }
}

// ---------------------------------------------------------------------------
// Launch: transpose weights -> LN1 -> QKV gemm -> attention -> proj(+res) ->
//         LN2 -> FC(+gelu) -> proj2(+res) -> d_out
// ---------------------------------------------------------------------------
extern "C" void kernel_launch(void* const* d_in, const int* in_sizes, int n_in,
                              void* d_out, int out_size, void* d_ws, size_t ws_size,
                              hipStream_t stream) {
  (void)in_sizes; (void)n_in; (void)out_size; (void)ws_size;
  const float* x        = (const float*)d_in[0];
  const float* ln1_g    = (const float*)d_in[1];
  const float* ln1_b    = (const float*)d_in[2];
  const float* c_attn_w = (const float*)d_in[3];
  const float* c_attn_b = (const float*)d_in[4];
  const float* c_proj_w = (const float*)d_in[5];
  const float* c_proj_b = (const float*)d_in[6];
  const float* ln2_g    = (const float*)d_in[7];
  const float* ln2_b    = (const float*)d_in[8];
  const float* fc_w     = (const float*)d_in[9];
  const float* fc_b     = (const float*)d_in[10];
  const float* proj2_w  = (const float*)d_in[11];
  const float* proj2_b  = (const float*)d_in[12];
  float* out = (float*)d_out;
  char* ws = (char*)d_ws;

  size_t off = 0;
  auto alloc = [&](size_t bytes) {
    size_t o = off;
    off += (bytes + 255) & ~(size_t)255;
    return o;
  };
  // Transposed f16 weights (N x K row-major)
  _Float16* wAttnT = (_Float16*)(ws + alloc((size_t)QKV_N * EMB * 2));
  _Float16* wProjT = (_Float16*)(ws + alloc((size_t)EMB * EMB * 2));
  _Float16* wFcT   = (_Float16*)(ws + alloc((size_t)4 * EMB * EMB * 2));
  _Float16* wP2T   = (_Float16*)(ws + alloc((size_t)EMB * 4 * EMB * 2));
  _Float16* bufA   = (_Float16*)(ws + alloc((size_t)NTOK * EMB * 2));      // h1 / attn out / h2
  _Float16* bufB   = (_Float16*)(ws + alloc((size_t)NTOK * 4 * EMB * 2));  // qkv, later ff
  float*    bufC   = (float*)   (ws + alloc((size_t)NTOK * EMB * 4));      // x1

  // W^T prep (K x N f32 -> N x K f16)
  cvt_transpose_f16<<<dim3(QKV_N / 32, EMB / 32), 256, 0, stream>>>(
      c_attn_w, wAttnT, EMB, QKV_N);
  cvt_transpose_f16<<<dim3(EMB / 32, EMB / 32), 256, 0, stream>>>(
      c_proj_w, wProjT, EMB, EMB);
  cvt_transpose_f16<<<dim3(4 * EMB / 32, EMB / 32), 256, 0, stream>>>(
      fc_w, wFcT, EMB, 4 * EMB);
  cvt_transpose_f16<<<dim3(EMB / 32, 4 * EMB / 32), 256, 0, stream>>>(
      proj2_w, wP2T, 4 * EMB, EMB);

  // h1 = LN1(x)
  ln_kernel<<<NTOK, 256, 0, stream>>>(x, ln1_g, ln1_b, bufA);
  // qkv = h1 @ c_attn_w + c_attn_b  (f16 out)
  gemm_f16<<<dim3(QKV_N / BN, NTOK / BM), 256, 0, stream>>>(
      bufA, wAttnT, c_attn_b, nullptr, nullptr, bufB, NTOK, QKV_N, EMB, 0);
  // a = sliding-window attention (f16 out, overwrites h1)
  attn_kernel<<<dim3(S_LEN / 64, NH, 2), 128, 0, stream>>>(bufB, bufA);
  // x1 = x + a @ c_proj_w + c_proj_b  (f32 out)
  gemm_f16<<<dim3(EMB / BN, NTOK / BM), 256, 0, stream>>>(
      bufA, wProjT, c_proj_b, x, bufC, nullptr, NTOK, EMB, EMB, 0);
  // h2 = LN2(x1)
  ln_kernel<<<NTOK, 256, 0, stream>>>(bufC, ln2_g, ln2_b, bufA);
  // ff = gelu(h2 @ fc_w + fc_b)  (f16 out, overwrites qkv)
  gemm_f16<<<dim3(4 * EMB / BN, NTOK / BM), 256, 0, stream>>>(
      bufA, wFcT, fc_b, nullptr, nullptr, bufB, NTOK, 4 * EMB, EMB, 1);
  // out = x1 + ff @ proj2_w + proj2_b  (f32 out)
  gemm_f16<<<dim3(EMB / BN, NTOK / BM), 256, 0, stream>>>(
      bufB, wP2T, proj2_b, bufC, out, nullptr, NTOK, EMB, 4 * EMB, 0);
}
